// GCN2_MSE_19834158972972
// MI455X (gfx1250) — matrix operand order repaired and checked by reading, not compiled, not run on hardware
//
#include <hip/hip_runtime.h>
#include <hip/hip_bf16.h>
#include <math.h>

// ---------------------------------------------------------------------------
// GCN (2x GCNConv + BN + ReLU) -> FC -> FC -> GEMV softplus, for gfx1250.
// GEMMs use v_wmma_f32_16x16x32_bf16. W is staged TRANSPOSED in LDS with a
// padded pitch so every A/B fragment is two ds_load_b128. Epilogue uses a
// block-uniform full-tile fast path: immediate-offset stores, no per-element
// predication. Edge aggregation is L2-resident (51 MB << 192 MB L2).
// ---------------------------------------------------------------------------

typedef __attribute__((ext_vector_type(16))) __bf16 v16bf;
typedef __attribute__((ext_vector_type(8)))  __bf16 v8bf;
typedef __attribute__((ext_vector_type(8)))  float  v8f;

union AFrag { v16bf v; v8bf h[2]; __bf16 e[16]; };
union CFrag { v8f  v; float  e[8];  };

#define HIDC   128
#define LPITCH 136          // LDS pitch in bf16 (272 B = 68 dwords; bank skew)
#define BN_EPS 1e-5f

// ---------------------------------------------------------------------------
// Degree / norm kernels
// ---------------------------------------------------------------------------
__global__ __launch_bounds__(256) void deg_init_kernel(float* __restrict__ deg, int n) {
    int i = blockIdx.x * 256 + threadIdx.x;
    if (i < n) deg[i] = 1.0f;   // self-loop contributes 1 to every node
}

__global__ __launch_bounds__(256) void deg_edges_kernel(const int* __restrict__ dst,
                                                        float* __restrict__ deg,
                                                        long long ne) {
    long long i = (long long)blockIdx.x * 256 + threadIdx.x;
    if (i < ne) atomicAdd(&deg[dst[i]], 1.0f);
}

__global__ __launch_bounds__(256) void deg_to_dinv_kernel(float* __restrict__ deg, int n) {
    int i = blockIdx.x * 256 + threadIdx.x;
    if (i < n) deg[i] = rsqrtf(deg[i]);   // deg >= 1 always (self-loop)
}

// ---------------------------------------------------------------------------
// WMMA GEMM: out[nrows x 128] = X[nrows x 128] @ W[128 x 128] (+ epilogue)
// MODE 0: plain store   MODE 1: +bias, BN, ReLU   MODE 2: +bias, ReLU
// Block: 256 threads (8 waves), tile = 128 rows x 128 cols, K = 128.
// sA  : X tile, row-major  [row][k],  pitch LPITCH
// sWt : W TRANSPOSED       [col][k],  pitch LPITCH  -> B frags contiguous in K
// ---------------------------------------------------------------------------
template <int MODE>
__global__ __launch_bounds__(256)
void gemm128_wmma_kernel(const float* __restrict__ X, const float* __restrict__ W,
                         float* __restrict__ out, int nrows,
                         const float* __restrict__ bias,
                         const float* __restrict__ gamma, const float* __restrict__ beta,
                         const float* __restrict__ mean,  const float* __restrict__ var) {
    __shared__ __bf16 sA[128 * LPITCH];    // ~34 KB
    __shared__ __bf16 sWt[128 * LPITCH];   // ~34 KB (transposed weights)

    const int t = threadIdx.x;
    const int rowBlock = blockIdx.x * 128;
    const bool full = (rowBlock + 128) <= nrows;   // block-uniform

    // Cooperative staging: 4096 float4 groups per matrix, 16 per thread.
#pragma unroll
    for (int it = 0; it < 16; ++it) {
        int idx = t + it * 256;            // float4 index
        int r = idx >> 5;                  // 32 float4 per 128-wide row
        int c = (idx & 31) * 4;
        // W row-major [k][o] -> sWt[o][k] (transposed, 4 scattered b16 stores)
        float4 w4 = *reinterpret_cast<const float4*>(W + r * 128 + c);
        sWt[(c + 0) * LPITCH + r] = (__bf16)w4.x;
        sWt[(c + 1) * LPITCH + r] = (__bf16)w4.y;
        sWt[(c + 2) * LPITCH + r] = (__bf16)w4.z;
        sWt[(c + 3) * LPITCH + r] = (__bf16)w4.w;
        // X tile -> sA row-major (contiguous 4xbf16 store)
        float4 x4 = make_float4(0.f, 0.f, 0.f, 0.f);
        if (full || (rowBlock + r) < nrows)
            x4 = *reinterpret_cast<const float4*>(X + (size_t)(rowBlock + r) * 128 + c);
        sA[r * LPITCH + c + 0] = (__bf16)x4.x;
        sA[r * LPITCH + c + 1] = (__bf16)x4.y;
        sA[r * LPITCH + c + 2] = (__bf16)x4.z;
        sA[r * LPITCH + c + 3] = (__bf16)x4.w;
    }
    __syncthreads();

    const int wave = t >> 5;
    const int lane = t & 31;
    const int half = lane >> 4;          // A: K+8 select; B: K+8 select
    const int nlo  = lane & 15;          // A: row M; B/D: column N
    const int aRow = wave * 16 + nlo;    // A row within the 128-row tile
    const int rbase = rowBlock + wave * 16 + half * 8;   // first D row (r=0)

    // A fragments for all four K=32 steps (ISA 16-bit A 16x32 layout):
    // e[0..7]  = K = ks*32 + half*8 + (0..7)
    // e[8..15] = K = ks*32 + 16 + half*8 + (0..7)   -> two b128 loads each
    AFrag afrag[4];
#pragma unroll
    for (int ks = 0; ks < 4; ++ks) {
        const __bf16* base = &sA[aRow * LPITCH + ks * 32 + half * 8];
        afrag[ks].h[0] = *reinterpret_cast<const v8bf*>(base);
        afrag[ks].h[1] = *reinterpret_cast<const v8bf*>(base + 16);
    }

#pragma unroll
    for (int ct = 0; ct < 8; ++ct) {        // 8 column tiles of 16
        CFrag acc;
#pragma unroll
        for (int r = 0; r < 8; ++r) acc.e[r] = 0.0f;

        const int col = ct * 16 + nlo;
#pragma unroll
        for (int ks = 0; ks < 4; ++ks) {
            // B 32x16: e[0..7] K=half*8+(0..7), e[8..15] K=16+half*8+(0..7) (matches A)
            AFrag bfrag;
            const __bf16* base = &sWt[col * LPITCH + ks * 32 + half * 8];
            bfrag.h[0] = *reinterpret_cast<const v8bf*>(base);
            bfrag.h[1] = *reinterpret_cast<const v8bf*>(base + 16);
            acc.v = __builtin_amdgcn_wmma_f32_16x16x32_bf16(
                false, afrag[ks].v, false, bfrag.v, (short)0, acc.v, false, false);
        }

        // Per-column epilogue constants (folded BN: v*scl + sft, then ReLU).
        float scl = 1.0f, sft = 0.0f;
        if (MODE == 1) {
            float s = gamma[col] * rsqrtf(var[col] + BN_EPS);
            scl = s;
            sft = (bias[col] - mean[col]) * s + beta[col];
        } else if (MODE == 2) {
            sft = bias[col];
        }

        if (full) {
            // Fast path: one base pointer, stores at immediate offsets r*512B.
            float* p = out + (size_t)rbase * 128 + col;
#pragma unroll
            for (int r = 0; r < 8; ++r) {   // D: M = rbase + r, N = col
                float v = acc.e[r];
                if (MODE == 1)      v = fmaxf(fmaf(v, scl, sft), 0.0f);
                else if (MODE == 2) v = fmaxf(v + sft, 0.0f);
                p[r * 128] = v;
            }
        } else {
#pragma unroll
            for (int r = 0; r < 8; ++r) {
                int grow = rbase + r;
                if (grow < nrows) {
                    float v = acc.e[r];
                    if (MODE == 1)      v = fmaxf(fmaf(v, scl, sft), 0.0f);
                    else if (MODE == 2) v = fmaxf(v + sft, 0.0f);
                    out[(size_t)grow * 128 + col] = v;
                }
            }
        }
    }
}

// ---------------------------------------------------------------------------
// Self-loop term (also initializes the scatter buffer): out = in * dinv[i]^2
// ---------------------------------------------------------------------------
__global__ __launch_bounds__(256)
void self_loop_kernel(const float* __restrict__ hin, const float* __restrict__ dinv,
                      float* __restrict__ hout, long long total) {
    long long i = (long long)blockIdx.x * 256 + threadIdx.x;
    if (i < total) {
        int node = (int)(i >> 7);
        float w = dinv[node];
        hout[i] = hin[i] * w * w;
    }
}

// ---------------------------------------------------------------------------
// Edge scatter: one wave per edge, float4 per lane, atomic f32 adds.
// Gather + scatter traffic stays in the 192 MB L2 (features = 51.2 MB).
// ---------------------------------------------------------------------------
__global__ __launch_bounds__(256)
void edge_scatter_kernel(const int* __restrict__ src, const int* __restrict__ dst,
                         const float* __restrict__ dinv, const float* __restrict__ hin,
                         float* __restrict__ hout, long long ne) {
    long long gid = (long long)blockIdx.x * 256 + threadIdx.x;
    long long e = gid >> 5;
    if (e >= ne) return;
    int lane = threadIdx.x & 31;
    int s = src[e];
    int d = dst[e];
    float w = dinv[s] * dinv[d];
    const float4 v = *reinterpret_cast<const float4*>(hin + (size_t)s * 128 + lane * 4);
    float* o = hout + (size_t)d * 128 + lane * 4;
    atomicAdd(o + 0, v.x * w);
    atomicAdd(o + 1, v.y * w);
    atomicAdd(o + 2, v.z * w);
    atomicAdd(o + 3, v.w * w);
}

// ---------------------------------------------------------------------------
// Post-aggregation fused epilogue: h = relu(bn(h + bias))
// ---------------------------------------------------------------------------
__global__ __launch_bounds__(256)
void bias_bn_relu_kernel(float* __restrict__ h, const float* __restrict__ bias,
                         const float* __restrict__ gamma, const float* __restrict__ beta,
                         const float* __restrict__ mean,  const float* __restrict__ var,
                         long long total) {
    long long i = (long long)blockIdx.x * 256 + threadIdx.x;
    if (i < total) {
        int c = (int)(i & 127);
        float s = gamma[c] * rsqrtf(var[c] + BN_EPS);
        float x = fmaf(h[i] + bias[c] - mean[c], s, beta[c]);
        h[i] = fmaxf(x, 0.0f);
    }
}

// ---------------------------------------------------------------------------
// GEMV 128 -> 1 with softplus: one wave per row, shuffle reduction.
// ---------------------------------------------------------------------------
__global__ __launch_bounds__(256)
void gemv_softplus_kernel(const float* __restrict__ H, const float* __restrict__ Wo,
                          const float* __restrict__ bo, float* __restrict__ out, int n) {
    long long gid = (long long)blockIdx.x * 256 + threadIdx.x;
    int row = (int)(gid >> 5);
    if (row >= n) return;
    int lane = threadIdx.x & 31;
    const float4 h4 = *reinterpret_cast<const float4*>(H + (size_t)row * 128 + lane * 4);
    const float4 w4 = *reinterpret_cast<const float4*>(Wo + lane * 4);
    float s = h4.x * w4.x + h4.y * w4.y + h4.z * w4.z + h4.w * w4.w;
#pragma unroll
    for (int off = 16; off > 0; off >>= 1) s += __shfl_down(s, off, 32);
    if (lane == 0) {
        float x = s + bo[0];
        out[row] = fmaxf(x, 0.0f) + log1pf(expf(-fabsf(x)));   // stable softplus
    }
}

// ---------------------------------------------------------------------------
// Driver
// ---------------------------------------------------------------------------
extern "C" void kernel_launch(void* const* d_in, const int* in_sizes, int n_in,
                              void* d_out, int out_size, void* d_ws, size_t ws_size,
                              hipStream_t stream) {
    (void)n_in; (void)out_size; (void)ws_size;

    const float* x   = (const float*)d_in[0];
    const int*   ei  = (const int*)d_in[1];      // [2, E] edge_index
    const float* W1  = (const float*)d_in[2];
    const float* b1  = (const float*)d_in[3];
    const float* W2  = (const float*)d_in[4];
    const float* b2  = (const float*)d_in[5];
    const float* Wf1 = (const float*)d_in[6];
    const float* bf1 = (const float*)d_in[7];
    const float* Wf2 = (const float*)d_in[8];
    const float* bf2 = (const float*)d_in[9];
    const float* Wo  = (const float*)d_in[10];
    const float* bo  = (const float*)d_in[11];
    const float* g1  = (const float*)d_in[12];
    const float* be1 = (const float*)d_in[13];
    const float* m1  = (const float*)d_in[14];
    const float* v1  = (const float*)d_in[15];
    const float* g2  = (const float*)d_in[16];
    const float* be2 = (const float*)d_in[17];
    const float* m2  = (const float*)d_in[18];
    const float* v2  = (const float*)d_in[19];
    const float* g3  = (const float*)d_in[20];
    const float* be3 = (const float*)d_in[21];
    const float* m3  = (const float*)d_in[22];
    const float* v3  = (const float*)d_in[23];

    const int       n  = in_sizes[0] / HIDC;      // 100000
    const long long E  = (long long)in_sizes[1] / 2;
    const int*      src = ei;
    const int*      dst = ei + E;

    // Workspace: two N x 128 f32 ping-pong buffers + dinv.
    float* bufA = (float*)d_ws;
    float* bufB = bufA + (size_t)n * HIDC;
    float* dinv = bufB + (size_t)n * HIDC;

    float* outPred = (float*)d_out;               // [N]
    float* outH1   = outPred + n;                 // [N, 128]

    const long long tot        = (long long)n * HIDC;
    const int nodeBlocks       = (n + 255) / 256;
    const int elemBlocks       = (int)((tot + 255) / 256);
    const int gemmBlocks       = (n + 127) / 128;
    const int edgeThreadBlocks = (int)((E + 255) / 256);
    const int edgeWaveBlocks   = (int)((E * 32 + 255) / 256);

    // 1) Symmetric norm: deg (with self-loop) -> dinv = rsqrt(deg)
    deg_init_kernel<<<nodeBlocks, 256, 0, stream>>>(dinv, n);
    deg_edges_kernel<<<edgeThreadBlocks, 256, 0, stream>>>(dst, dinv, E);
    deg_to_dinv_kernel<<<nodeBlocks, 256, 0, stream>>>(dinv, n);

    // 2) GCN layer 1: h = x @ W1; agg; +b1, BN1, ReLU
    gemm128_wmma_kernel<0><<<gemmBlocks, 256, 0, stream>>>(
        x, W1, bufA, n, nullptr, nullptr, nullptr, nullptr, nullptr);
    self_loop_kernel<<<elemBlocks, 256, 0, stream>>>(bufA, dinv, bufB, tot);
    edge_scatter_kernel<<<edgeWaveBlocks, 256, 0, stream>>>(src, dst, dinv, bufA, bufB, E);
    bias_bn_relu_kernel<<<elemBlocks, 256, 0, stream>>>(bufB, b1, g1, be1, m1, v1, tot);

    // 3) GCN layer 2
    gemm128_wmma_kernel<0><<<gemmBlocks, 256, 0, stream>>>(
        bufB, W2, bufA, n, nullptr, nullptr, nullptr, nullptr, nullptr);
    self_loop_kernel<<<elemBlocks, 256, 0, stream>>>(bufA, dinv, bufB, tot);
    edge_scatter_kernel<<<edgeWaveBlocks, 256, 0, stream>>>(src, dst, dinv, bufA, bufB, E);
    bias_bn_relu_kernel<<<elemBlocks, 256, 0, stream>>>(bufB, b2, g2, be2, m2, v2, tot);

    // 4) FC1: relu(bn3(h @ Wf1 + bf1)) — fused epilogue
    gemm128_wmma_kernel<1><<<gemmBlocks, 256, 0, stream>>>(
        bufB, Wf1, bufA, n, bf1, g3, be3, m3, v3);

    // 5) FC2: h1 = relu(h @ Wf2 + bf2) — written directly into d_out
    gemm128_wmma_kernel<2><<<gemmBlocks, 256, 0, stream>>>(
        bufA, Wf2, outH1, n, bf2, nullptr, nullptr, nullptr, nullptr);

    // 6) raw_pred = softplus(h1 @ Wo + bo)
    gemv_softplus_kernel<<<(int)(((long long)n * 32 + 255) / 256), 256, 0, stream>>>(
        outH1, Wo, bo, outPred, n);
}
